// DeepEmbedAttention_57320633532866
// MI455X (gfx1250) — compile-verified, matrix operand order checked by reading, not run on hardware
//
#include <hip/hip_runtime.h>
#include <hip/hip_bf16.h>
#include <math.h>

// ---------------------------------------------------------------------------
// DeepEmbedAttention forward for gfx1250 (MI455X), wave32 + WMMA f16 + TDM.
// Pipeline:
//   cvt4_kernel     : x / W_qq / W_k / W_v fp32 -> f16 (WMMA operands)
//   proj_kernel     : qraw = x@Wqq^T, kdraw = x@Wk^T, vdraw = x@Wv^T
//                     (x tile staged into LDS by TENSOR_LOAD_TO_LDS w/ padding)
//   up_kernel       : kraw = (kd@Wkup^T)*k_emb ; vraw = tanh(vd@Wvup^T)*v_emb
//   shift_ln_kernel : time-shift mix + LayerNorm; f16 qh, kh, vth ([B,C,T])
//   attn_kernel     : flash attention, cap-tanh, causal, online softmax (WMMA;
//                     q tile staged by TDM)
// ---------------------------------------------------------------------------

#define B_    4
#define T_    2048
#define C_    1024
#define QD_   256
#define KVD_  32

typedef _Float16     v16h  __attribute__((ext_vector_type(16)));
typedef _Float16     half8 __attribute__((ext_vector_type(8)));
typedef float        v8f   __attribute__((ext_vector_type(8)));
typedef unsigned int u32x4 __attribute__((ext_vector_type(4)));
typedef int          i32x8 __attribute__((ext_vector_type(8)));
typedef int          i32x4 __attribute__((ext_vector_type(4)));

// Assemble a 16x32 f16 A-style fragment (ISA 7.12.2 layout) from a lane-local
// pointer: 8 contiguous halfs at p, 8 contiguous halfs at p+16.
static __device__ __forceinline__ v16h make_frag16(const _Float16* p) {
    half8 lo = *(const half8*)(p);
    half8 hh = *(const half8*)(p + 16);
    return __builtin_shufflevector(lo, hh, 0, 1, 2, 3, 4, 5, 6, 7,
                                           8, 9, 10, 11, 12, 13, 14, 15);
}

static __device__ __forceinline__ v8f wmma_f16(v16h a, v16h b, v8f c) {
    return __builtin_amdgcn_wmma_f32_16x16x32_f16(false, a, false, b,
                                                  (short)0, c, false, false);
}

// ---------------------------------------------------------------------------
// Tensor Data Mover: 2-D tile (tile1 rows x tile0 elements, 2-byte elements)
// from global row-major [tdim1, stride0] into LDS at lds_off, with optional
// LDS padding (ISA 8.4: pad every (2<<pad_interval) DWORDs by (pad_amount+1)
// DWORDs).  Issue from ONE wave; completion via s_wait_tensorcnt.
// NOTE: this toolchain exposes the 6-arg builtin form
//   (u32x4 g0, i32x8 g1, i32x4 g2, i32x4 g3, i32x8 g4, i32 cpol).
static __device__ __forceinline__ void tdm_load_2d_f16(
    unsigned int lds_off, const void* gptr,
    unsigned int tdim0, unsigned int tdim1,
    unsigned int tile0, unsigned int tile1, unsigned int stride0,
    unsigned int pad_interval_code, unsigned int pad_amount_code,
    unsigned int pad_enable)
{
    unsigned long long ga = (unsigned long long)(size_t)gptr;
    u32x4 g0;
    g0[0] = 1u;                                   // count=1, user descriptor
    g0[1] = lds_off;                              // lds_addr
    g0[2] = (unsigned int)ga;                     // global_addr[31:0]
    g0[3] = (unsigned int)((ga >> 32) & 0x1FFFFFFu) | 0x80000000u; // addr hi | type=2
    i32x8 g1;
    g1[0] = (int)((1u << 16) |                    // data_size = 2 bytes
                  (pad_enable << 20) |
                  (pad_interval_code << 22) |
                  (pad_amount_code << 25));
    g1[1] = (int)(tdim0 << 16);                   // tensor_dim0[15:0]
    g1[2] = (int)((tdim0 >> 16) | (tdim1 << 16)); // tensor_dim0 hi | dim1 lo
    g1[3] = (int)((tdim1 >> 16) | (tile0 << 16)); // tensor_dim1 hi | tile_dim0
    g1[4] = (int)(tile1 & 0xFFFFu);               // tile_dim1 (tile_dim2 = 0)
    g1[5] = (int)stride0;                         // tensor_dim0_stride[31:0]
    g1[6] = 0;
    g1[7] = 0;
    i32x4 z4;
    z4[0] = 0; z4[1] = 0; z4[2] = 0; z4[3] = 0;   // 2-D: groups 2/3 unused
    i32x8 z8;
    z8[0] = 0; z8[1] = 0; z8[2] = 0; z8[3] = 0;
    z8[4] = 0; z8[5] = 0; z8[6] = 0; z8[7] = 0;
    __builtin_amdgcn_tensor_load_to_lds(g0, g1, z4, z4, z8, 0);
}

// ---------------------------------------------------------------------------
__global__ void cvt4_kernel(const float* __restrict__ src,
                            _Float16* __restrict__ dst, int n4) {
    int i = blockIdx.x * 256 + threadIdx.x;
    if (i < n4) {
        float4 v = *(const float4*)(src + (size_t)i * 4);
        _Float16* d = dst + (size_t)i * 4;
        d[0] = (_Float16)v.x; d[1] = (_Float16)v.y;
        d[2] = (_Float16)v.z; d[3] = (_Float16)v.w;
    }
}

// ---------------------------------------------------------------------------
// proj_kernel: one block = 16 rows of x.  8 waves.
//   waves 0..7 : q columns [32w, 32w+32)  (two 16x16 WMMA tiles each)
//   waves 0..3 : kd / vd (32 columns each of W_k / W_v)
// x tile (16 x 1024 f16) DMA'd into LDS by the TDM.  Row = 2048B data; pad of
// 8B every 1024B streamed -> row stride 1032 halfs, 4-half hole after col 512.
#define XPAD 1032

__global__ __launch_bounds__(256) void proj_kernel(
    const _Float16* __restrict__ xh,   // [B*T,1024] f16
    const _Float16* __restrict__ Wq,   // [256,1024] f16
    const _Float16* __restrict__ Wk,   // [32,1024]  f16
    const _Float16* __restrict__ Wv,   // [32,1024]  f16
    float* __restrict__ qraw,          // [B*T,256]
    float* __restrict__ kdraw,         // [B*T,32]
    float* __restrict__ vdraw)         // [B*T,32]
{
    __shared__ __align__(16) _Float16 xs[16 * XPAD];

    const int tid  = threadIdx.x;
    const int lane = tid & 31;
    const int w    = tid >> 5;
    const int lr   = lane & 15;
    const int hi   = lane >> 4;
    const int bt0  = blockIdx.x * 16;

    if (w == 0) {
        // 16 x 1024 f16 tile; pad_interval code 7 (256 DW), pad_amount code 1 (2 DW)
        tdm_load_2d_f16((unsigned int)(size_t)&xs[0],
                        xh + (size_t)bt0 * C_,
                        /*tdim0=*/C_, /*tdim1=*/B_ * T_,
                        /*tile0=*/C_, /*tile1=*/16, /*stride0=*/C_,
                        /*pad_interval=*/7, /*pad_amount=*/1, /*pad_enable=*/1);
        __builtin_amdgcn_s_wait_tensorcnt(0);
    }
    __syncthreads();

    const _Float16* ap = xs + lr * XPAD + hi * 8;

    // ---- q projection ----
    {
        v8f a0, a1;
        #pragma unroll
        for (int r = 0; r < 8; ++r) { a0[r] = 0.f; a1[r] = 0.f; }
        const int n0 = (2 * w) * 16 + lr;
        const int n1 = (2 * w + 1) * 16 + lr;
        const _Float16* b0p = Wq + (size_t)n0 * C_ + hi * 8;
        const _Float16* b1p = Wq + (size_t)n1 * C_ + hi * 8;
        #pragma unroll 4
        for (int k0 = 0; k0 < C_; k0 += 32) {
            const int adj = (k0 >= 512) ? 4 : 0;   // skip mid-row pad hole
            v16h a  = make_frag16(ap + k0 + adj);
            v16h f0 = make_frag16(b0p + k0);
            v16h f1 = make_frag16(b1p + k0);
            a0 = wmma_f16(a, f0, a0);
            a1 = wmma_f16(a, f1, a1);
        }
        #pragma unroll
        for (int r = 0; r < 8; ++r) {
            int row = r + 8 * hi;
            qraw[(size_t)(bt0 + row) * QD_ + (2 * w) * 16 + lr]     = a0[r];
            qraw[(size_t)(bt0 + row) * QD_ + (2 * w + 1) * 16 + lr] = a1[r];
        }
    }

    // ---- kd / vd projections (waves 0..3) ----
    if (w < 4) {
        const bool isK = (w < 2);
        const _Float16* Ws = isK ? Wk : Wv;
        float* dst = isK ? kdraw : vdraw;
        const int nc0 = (isK ? w : (w - 2)) * 16;
        v8f acc;
        #pragma unroll
        for (int r = 0; r < 8; ++r) acc[r] = 0.f;
        const _Float16* bp = Ws + (size_t)(nc0 + lr) * C_ + hi * 8;
        #pragma unroll 4
        for (int k0 = 0; k0 < C_; k0 += 32) {
            const int adj = (k0 >= 512) ? 4 : 0;
            v16h a = make_frag16(ap + k0 + adj);
            v16h f = make_frag16(bp + k0);
            acc = wmma_f16(a, f, acc);
        }
        #pragma unroll
        for (int r = 0; r < 8; ++r)
            dst[(size_t)(bt0 + r + 8 * hi) * KVD_ + nc0 + lr] = acc[r];
    }
}

// ---------------------------------------------------------------------------
// up_kernel: one block per token.  K=32 up-projections + embedding gathers.
__global__ __launch_bounds__(256) void up_kernel(
    const float* __restrict__ kdraw, const float* __restrict__ vdraw,
    const float* __restrict__ Wkup,  // [256,32]
    const float* __restrict__ Wvup,  // [1024,32]
    const int* __restrict__ idx,
    const float* __restrict__ ktab,  // [VOCAB,256]
    const float* __restrict__ vtab,  // [VOCAB,1024]
    float* __restrict__ kraw, float* __restrict__ vraw)
{
    __shared__ float kd[KVD_], vd[KVD_];
    const int bt = blockIdx.x, tid = threadIdx.x;
    if (tid < KVD_) {
        kd[tid] = kdraw[(size_t)bt * KVD_ + tid];
        vd[tid] = vdraw[(size_t)bt * KVD_ + tid];
    }
    __syncthreads();
    const int tok = idx[bt];
    {
        float s = 0.f;
        #pragma unroll
        for (int d = 0; d < KVD_; ++d) s += kd[d] * Wkup[tid * KVD_ + d];
        kraw[(size_t)bt * QD_ + tid] = s * ktab[(size_t)tok * QD_ + tid];
    }
    #pragma unroll
    for (int j = 0; j < 4; ++j) {
        int c = tid + j * 256;
        float s = 0.f;
        #pragma unroll
        for (int d = 0; d < KVD_; ++d) s += vd[d] * Wvup[c * KVD_ + d];
        vraw[(size_t)bt * C_ + c] = tanhf(s) * vtab[(size_t)tok * C_ + c];
    }
}

// ---------------------------------------------------------------------------
static __device__ __forceinline__ float block_sum256(float v, float* sbuf) {
    const int tid = threadIdx.x, lane = tid & 31, wid = tid >> 5;
    #pragma unroll
    for (int off = 16; off > 0; off >>= 1) v += __shfl_down(v, off, 32);
    __syncthreads();
    if (lane == 0) sbuf[wid] = v;
    __syncthreads();
    if (tid == 0) {
        float s = 0.f;
        #pragma unroll
        for (int i = 0; i < 8; ++i) s += sbuf[i];
        sbuf[0] = s;
    }
    __syncthreads();
    return sbuf[0];
}

// shift_ln_kernel: one block per token; time-shift mix + LayerNorm; f16 out.
__global__ __launch_bounds__(256) void shift_ln_kernel(
    const float* __restrict__ qraw, const float* __restrict__ kraw,
    const float* __restrict__ vraw,
    const float* __restrict__ xq, const float* __restrict__ xk,
    const float* __restrict__ xv,
    const float* __restrict__ gq, const float* __restrict__ bq,
    const float* __restrict__ gk, const float* __restrict__ bk,
    const float* __restrict__ gv, const float* __restrict__ bv,
    _Float16* __restrict__ qh, _Float16* __restrict__ kh,
    _Float16* __restrict__ vth)     // vth: [B, C, T] transposed
{
    __shared__ float red[8];
    const int bt = blockIdx.x, tid = threadIdx.x;
    const int t = bt & (T_ - 1), b = bt >> 11;
    const bool hasPrev = (t > 0);

    { // q
        float cur  = qraw[(size_t)bt * QD_ + tid];
        float prev = hasPrev ? qraw[(size_t)(bt - 1) * QD_ + tid] : 0.f;
        float m  = cur + (prev - cur) * xq[tid];
        float mu = block_sum256(m, red) * (1.f / QD_);
        float va = block_sum256(m * m, red) * (1.f / QD_) - mu * mu;
        float y  = (m - mu) * rsqrtf(va + 1e-5f) * gq[tid] + bq[tid];
        qh[(size_t)bt * QD_ + tid] = (_Float16)y;
    }
    { // k
        float cur  = kraw[(size_t)bt * QD_ + tid];
        float prev = hasPrev ? kraw[(size_t)(bt - 1) * QD_ + tid] : 0.f;
        float m  = cur + (prev - cur) * xk[tid];
        float mu = block_sum256(m, red) * (1.f / QD_);
        float va = block_sum256(m * m, red) * (1.f / QD_) - mu * mu;
        float y  = (m - mu) * rsqrtf(va + 1e-5f) * gk[tid] + bk[tid];
        kh[(size_t)bt * QD_ + tid] = (_Float16)y;
    }
    { // v (4 channels per thread), store transposed for PV B-fragments
        float m[4], s = 0.f, s2 = 0.f;
        #pragma unroll
        for (int j = 0; j < 4; ++j) {
            int c = tid + j * 256;
            float cur  = vraw[(size_t)bt * C_ + c];
            float prev = hasPrev ? vraw[(size_t)(bt - 1) * C_ + c] : 0.f;
            m[j] = cur + (prev - cur) * xv[c];
            s += m[j]; s2 += m[j] * m[j];
        }
        float mu = block_sum256(s, red) * (1.f / C_);
        float va = block_sum256(s2, red) * (1.f / C_) - mu * mu;
        float ri = rsqrtf(va + 1e-5f);
        #pragma unroll
        for (int j = 0; j < 4; ++j) {
            int c = tid + j * 256;
            float y = (m[j] - mu) * ri * gv[c] + bv[c];
            vth[((size_t)(b * C_ + c)) * T_ + t] = (_Float16)y;
        }
    }
}

// ---------------------------------------------------------------------------
// attn_kernel: flash attention.  Block = (b, 16 query rows), 8 waves.
// S-blocks of 128 keys; wave w computes score tile for keys [s0+16w, s0+16w+16)
// and owns v channels [128w, 128w+128) (8 accumulation tiles).
// q tile DMA'd into LDS by TDM: row 512B + 16B pad -> stride 264 halfs.
#define QPAD 264   // halfs
#define SPAD 132   // floats
#define PPAD 136   // halfs

__global__ __launch_bounds__(256) void attn_kernel(
    const _Float16* __restrict__ qh, const _Float16* __restrict__ kh,
    const _Float16* __restrict__ vth, float* __restrict__ out)
{
    __shared__ __align__(16) _Float16 qs[16 * QPAD];
    __shared__ float    sc[16 * SPAD];
    __shared__ _Float16 ps[16 * PPAD];
    __shared__ float    mrow[16], lrow[16], crow[16];

    const int tid  = threadIdx.x;
    const int lane = tid & 31;
    const int w    = tid >> 5;
    const int lr   = lane & 15;
    const int hi   = lane >> 4;
    const int t0   = blockIdx.x * 16;
    const int b    = blockIdx.y;

    if (w == 0) {
        // 16 x 256 f16 tile; pad_interval code 6 (128 DW), pad_amount code 3 (4 DW)
        tdm_load_2d_f16((unsigned int)(size_t)&qs[0],
                        qh + (size_t)(b * T_ + t0) * QD_,
                        /*tdim0=*/QD_, /*tdim1=*/B_ * T_,
                        /*tile0=*/QD_, /*tile1=*/16, /*stride0=*/QD_,
                        /*pad_interval=*/6, /*pad_amount=*/3, /*pad_enable=*/1);
    }
    if (tid < 16) { mrow[tid] = -1e30f; lrow[tid] = 0.f; }
    if (w == 0) __builtin_amdgcn_s_wait_tensorcnt(0);
    __syncthreads();

    v8f acc[8];
    #pragma unroll
    for (int n = 0; n < 8; ++n)
        #pragma unroll
        for (int r = 0; r < 8; ++r) acc[n][r] = 0.f;

    const int nsb = (t0 + 15) / 128 + 1;
    for (int sb = 0; sb < nsb; ++sb) {
        const int s0  = sb * 128;
        const int key = s0 + w * 16 + lr;

        if (sb + 1 < nsb)  // pull next key block toward L2/WGP$
            __builtin_prefetch(kh + ((size_t)(b * T_ + key + 128)) * QD_, 0, 1);

        // ---- scores: q(16x256) . k_tile^T ----
        v8f s;
        #pragma unroll
        for (int r = 0; r < 8; ++r) s[r] = 0.f;
        const _Float16* qp = qs + lr * QPAD + hi * 8;
        const _Float16* kp = kh + ((size_t)(b * T_ + key)) * QD_ + hi * 8;
        #pragma unroll
        for (int k0 = 0; k0 < QD_; k0 += 32) {
            v16h a  = make_frag16(qp + k0);
            v16h bb = make_frag16(kp + k0);
            s = wmma_f16(a, bb, s);
        }
        #pragma unroll
        for (int r = 0; r < 8; ++r) {
            int rg = t0 + r + 8 * hi;
            float sv = 64.f * tanhf(s[r] * (1.f / 1024.f));
            if (key > rg) sv = -1e30f;               // causal mask
            sc[(r + 8 * hi) * SPAD + w * 16 + lr] = sv;
        }
        __syncthreads();

        // ---- online softmax (one thread per row) ----
        if (tid < 16) {
            const int row = tid;
            float rmax = -1e30f;
            for (int j = 0; j < 128; ++j) rmax = fmaxf(rmax, sc[row * SPAD + j]);
            if (rmax > -1e29f) {
                float m_old = mrow[row];
                float m_new = fmaxf(m_old, rmax);
                float corr  = __expf(m_old - m_new);
                float sum = 0.f;
                for (int j = 0; j < 128; ++j) {
                    float p = __expf(sc[row * SPAD + j] - m_new);
                    ps[row * PPAD + j] = (_Float16)p;
                    sum += p;
                }
                mrow[row] = m_new;
                lrow[row] = lrow[row] * corr + sum;
                crow[row] = corr;
            } else {
                for (int j = 0; j < 128; ++j) ps[row * PPAD + j] = (_Float16)0.f;
                crow[row] = 1.f;
            }
        }
        __syncthreads();

        // ---- rescale accumulators ----
        float cr[8];
        #pragma unroll
        for (int r = 0; r < 8; ++r) cr[r] = crow[r + 8 * hi];
        #pragma unroll
        for (int n = 0; n < 8; ++n)
            #pragma unroll
            for (int r = 0; r < 8; ++r) acc[n][r] *= cr[r];

        // ---- PV: P(16x128) . V(128 x 128-slice) ----
        #pragma unroll
        for (int kg = 0; kg < 4; ++kg) {
            v16h ap2 = make_frag16(ps + lr * PPAD + kg * 32 + hi * 8);
            #pragma unroll
            for (int n = 0; n < 8; ++n) {
                const int c = w * 128 + n * 16 + lr;
                const _Float16* vp =
                    vth + ((size_t)(b * C_ + c)) * T_ + s0 + kg * 32 + hi * 8;
                v16h bv = make_frag16(vp);
                acc[n] = wmma_f16(ap2, bv, acc[n]);
            }
        }
        __syncthreads();
    }

    // ---- epilogue: divide by softmax denominator, write fp32 out ----
    float linv[8];
    #pragma unroll
    for (int r = 0; r < 8; ++r) linv[r] = 1.f / lrow[r + 8 * hi];
    #pragma unroll
    for (int n = 0; n < 8; ++n)
        #pragma unroll
        for (int r = 0; r < 8; ++r) {
            int row = r + 8 * hi;
            int c   = w * 128 + n * 16 + lr;
            out[((size_t)(b * T_ + t0 + row)) * C_ + c] = acc[n][r] * linv[r];
        }
}

// ---------------------------------------------------------------------------
extern "C" void kernel_launch(void* const* d_in, const int* in_sizes, int n_in,
                              void* d_out, int out_size, void* d_ws, size_t ws_size,
                              hipStream_t stream) {
    (void)in_sizes; (void)n_in; (void)out_size; (void)ws_size;

    const float* x     = (const float*)d_in[0];
    const int*   idx   = (const int*)d_in[1];
    const float* Wqq   = (const float*)d_in[2];
    const float* Wk    = (const float*)d_in[3];
    const float* Wkup  = (const float*)d_in[4];
    const float* Wv    = (const float*)d_in[5];
    const float* Wvup  = (const float*)d_in[6];
    const float* ktab  = (const float*)d_in[7];
    const float* vtab  = (const float*)d_in[8];
    const float* xq    = (const float*)d_in[9];
    const float* xk    = (const float*)d_in[10];
    const float* xv    = (const float*)d_in[11];
    const float* gq    = (const float*)d_in[12];
    const float* bq    = (const float*)d_in[13];
    const float* gk    = (const float*)d_in[14];
    const float* bk    = (const float*)d_in[15];
    const float* gv    = (const float*)d_in[16];
    const float* bv    = (const float*)d_in[17];
    float* out = (float*)d_out;

    char* ws = (char*)d_ws;
    float*    qraw  = (float*)(ws + 0);               //  8 MiB
    float*    kdraw = (float*)(ws + 8388608);         //  1 MiB
    float*    vdraw = (float*)(ws + 9437184);         //  1 MiB
    float*    kraw  = (float*)(ws + 10485760);        //  8 MiB
    float*    vraw  = (float*)(ws + 18874368);        // 32 MiB
    _Float16* qhb   = (_Float16*)(ws + 52428800);     //  4 MiB
    _Float16* khb   = (_Float16*)(ws + 56623104);     //  4 MiB
    _Float16* vthb  = (_Float16*)(ws + 60817408);     // 16 MiB
    _Float16* Wq_h  = (_Float16*)(ws + 77594624);     // 512 KiB
    _Float16* Wk_h  = (_Float16*)(ws + 78118912);     //  64 KiB
    _Float16* Wv_h  = (_Float16*)(ws + 78184448);     //  64 KiB
    _Float16* xhb   = (_Float16*)(ws + 78249984);     //  16 MiB

    cvt4_kernel<<<(QD_ * C_ / 4 + 255) / 256, 256, 0, stream>>>(Wqq, Wq_h, QD_ * C_ / 4);
    cvt4_kernel<<<(KVD_ * C_ / 4 + 255) / 256, 256, 0, stream>>>(Wk, Wk_h, KVD_ * C_ / 4);
    cvt4_kernel<<<(KVD_ * C_ / 4 + 255) / 256, 256, 0, stream>>>(Wv, Wv_h, KVD_ * C_ / 4);
    cvt4_kernel<<<(B_ * T_ * C_ / 4 + 255) / 256, 256, 0, stream>>>(x, xhb, B_ * T_ * C_ / 4);

    proj_kernel<<<(B_ * T_) / 16, 256, 0, stream>>>(xhb, Wq_h, Wk_h, Wv_h,
                                                    qraw, kdraw, vdraw);

    up_kernel<<<B_ * T_, 256, 0, stream>>>(kdraw, vdraw, Wkup, Wvup, idx,
                                           ktab, vtab, kraw, vraw);

    shift_ln_kernel<<<B_ * T_, 256, 0, stream>>>(qraw, kraw, vraw,
                                                 xq, xk, xv, gq, bq, gk, bk,
                                                 gv, bv, qhb, khb, vthb);

    attn_kernel<<<dim3(T_ / 16, B_), 256, 0, stream>>>(qhb, khb, vthb, out);
}
